// JKNet_59072980189624
// MI455X (gfx1250) — compile-verified, matrix operand order checked by reading, not run on hardware
//
#include <hip/hip_runtime.h>
#include <hip/hip_bf16.h>

typedef __attribute__((ext_vector_type(16))) _Float16 v16h;
typedef __attribute__((ext_vector_type(8)))  float    v8f;

// ---------------------------------------------------------------------------
// Utility kernels
// ---------------------------------------------------------------------------
__global__ void zero_kernel(float* __restrict__ p, long long n) {
  long long i = (long long)blockIdx.x * blockDim.x + threadIdx.x;
  long long stride = (long long)gridDim.x * blockDim.x;
  for (; i < n; i += stride) p[i] = 0.0f;
}

__global__ void degree_kernel(const int* __restrict__ src, const int* __restrict__ dst,
                              float* __restrict__ deg_out, float* __restrict__ deg_in, int E) {
  int i = blockIdx.x * blockDim.x + threadIdx.x;
  if (i < E) {
    unsafeAtomicAdd(&deg_out[src[i]], 1.0f);   // global_atomic_add_f32
    unsafeAtomicAdd(&deg_in[dst[i]], 1.0f);
  }
}

// deg -> rsqrt(max(deg, 1)) in place (applied to both deg arrays at once)
__global__ void isqrt_kernel(float* __restrict__ d, long long n) {
  long long i = (long long)blockIdx.x * blockDim.x + threadIdx.x;
  if (i < n) d[i] = rsqrtf(fmaxf(d[i], 1.0f));
}

// ---------------------------------------------------------------------------
// Pre-pack a weight matrix W [K x ldw] (f32 row-major) into per-lane WMMA
// B-fragments (f16), fragment order [kt][ntile][lane][16 halves].
// B layout for v_wmma_f32_16x16x32_f16: lane half selects K block
// (lanes 0-15: K kb+0..7 & kb+16..23; lanes 16-31: K kb+8..15 & kb+24..31),
// column = ntile*16 + (lane & 15).
// One thread packs one lane-fragment (16 halves).
// ---------------------------------------------------------------------------
__global__ void pack_w_kernel(const float* __restrict__ W, int ldw,
                              int ktiles, int ntiles, _Float16* __restrict__ out) {
  int idx = blockIdx.x * blockDim.x + threadIdx.x;
  int total = ktiles * ntiles * 32;
  if (idx >= total) return;
  const int lane = idx & 31;
  const int t    = (idx >> 5) % ntiles;
  const int kt   = (idx >> 5) / ntiles;
  const int half = lane >> 4;
  const int r    = lane & 15;
  const int col  = t * 16 + r;
  const float* wp = W + (size_t)(kt * 32 + half * 8) * ldw + col;
  _Float16* o = out + (size_t)idx * 16;
#pragma unroll
  for (int i = 0; i < 8; ++i) o[i]     = (_Float16)wp[(size_t)i * ldw];
#pragma unroll
  for (int i = 0; i < 8; ++i) o[8 + i] = (_Float16)wp[(size_t)(16 + i) * ldw];
}

// ---------------------------------------------------------------------------
// Dense GEMM via v_wmma_f32_16x16x32_f16.
//   C[nrows x (NTILES*16)] = (A * scale[row]) @ W (+ bias)
// A: f32 row-major (converted to f16 on the fly, fused with deg^-1/2 scale).
// Wp: pre-packed f16 B-fragments (see pack_w_kernel) -> one 32-byte
// contiguous vector load per fragment, fully coalesced per wave.
// One wave handles a 16-row tile; NTILES accumulators cover the columns.
// ---------------------------------------------------------------------------
template <int NTILES>
__global__ __launch_bounds__(256)
void gemm_wmma_kernel(const float* __restrict__ A, int lda,
                      const float* __restrict__ scale,
                      const _Float16* __restrict__ Wp,
                      const float* __restrict__ bias,
                      float* __restrict__ C, int ldc,
                      int nrows, int ktiles) {
  const int wave = threadIdx.x >> 5;
  const int lane = threadIdx.x & 31;
  const int half = lane >> 4;        // 0 or 1
  const int r    = lane & 15;
  const int rowbase = (blockIdx.x * 8 + wave) * 16;
  if (rowbase >= nrows) return;      // uniform per wave -> EXEC stays all-ones

  v8f acc[NTILES];
#pragma unroll
  for (int t = 0; t < NTILES; ++t) acc[t] = (v8f){0,0,0,0,0,0,0,0};

  int arow = rowbase + r;
  if (arow >= nrows) arow = nrows - 1;          // clamp load, mask store later
  const float s = scale ? scale[arow] : 1.0f;
  const float* abase = A + (size_t)arow * lda + half * 8;
  const v16h* wfrag = (const v16h*)Wp;

  for (int kt = 0; kt < ktiles; ++kt) {
    const float* ap = abase + kt * 32;
    float4 f0 = *(const float4*)(ap + 0);       // K = kb+0 .. kb+3
    float4 f1 = *(const float4*)(ap + 4);       // K = kb+4 .. kb+7
    float4 f2 = *(const float4*)(ap + 16);      // K = kb+16 .. kb+19
    float4 f3 = *(const float4*)(ap + 20);      // K = kb+20 .. kb+23
    v16h a;
    a[0] = (_Float16)(f0.x * s); a[1] = (_Float16)(f0.y * s);
    a[2] = (_Float16)(f0.z * s); a[3] = (_Float16)(f0.w * s);
    a[4] = (_Float16)(f1.x * s); a[5] = (_Float16)(f1.y * s);
    a[6] = (_Float16)(f1.z * s); a[7] = (_Float16)(f1.w * s);
    a[8]  = (_Float16)(f2.x * s); a[9]  = (_Float16)(f2.y * s);
    a[10] = (_Float16)(f2.z * s); a[11] = (_Float16)(f2.w * s);
    a[12] = (_Float16)(f3.x * s); a[13] = (_Float16)(f3.y * s);
    a[14] = (_Float16)(f3.z * s); a[15] = (_Float16)(f3.w * s);

#pragma unroll
    for (int t = 0; t < NTILES; ++t) {
      const v16h b = wfrag[(size_t)(kt * NTILES + t) * 32 + lane];
      acc[t] = __builtin_amdgcn_wmma_f32_16x16x32_f16(
          false, a, false, b, (short)0, acc[t], false, false);
    }
  }

#pragma unroll
  for (int t = 0; t < NTILES; ++t) {
    const int col = t * 16 + r;
    const float bv = bias ? bias[col] : 0.0f;
#pragma unroll
    for (int v = 0; v < 8; ++v) {
      const int row = rowbase + half * 8 + v;   // C/D layout: M = half*8 + vgpr
      if (row < nrows) C[(size_t)row * ldc + col] = acc[t][v] + bv;
    }
  }
}

// ---------------------------------------------------------------------------
// Edge-parallel scatter-add:  agg[dst[e], :] += h[src[e], :]
// One thread handles 4 consecutive features (float4 gather hits L2-resident h).
// ---------------------------------------------------------------------------
__global__ void edge_agg_kernel(const float* __restrict__ h, int ldh, int fq,
                                const int* __restrict__ src, const int* __restrict__ dst,
                                float* __restrict__ agg, int ldagg, long long total) {
  long long i = (long long)blockIdx.x * blockDim.x + threadIdx.x;
  long long stride = (long long)gridDim.x * blockDim.x;
  for (; i < total; i += stride) {
    const int e = (int)(i / fq);
    const int f = (int)(i % fq) * 4;
    const int s = src[e];
    const int d = dst[e];
    const float4 v = *(const float4*)(h + (size_t)s * ldh + f);
    float* o = agg + (size_t)d * ldagg + f;
    unsafeAtomicAdd(o + 0, v.x);
    unsafeAtomicAdd(o + 1, v.y);
    unsafeAtomicAdd(o + 2, v.z);
    unsafeAtomicAdd(o + 3, v.w);
  }
}

// out[row, col@ldout] = relu(agg[row, col@64] * di[row] + bias[col]),  64 feats
__global__ void relu_scale_bias_kernel(const float* __restrict__ agg,
                                       const float* __restrict__ di,
                                       const float* __restrict__ bias,
                                       float* __restrict__ out, int ldout, int nrows) {
  int i = blockIdx.x * blockDim.x + threadIdx.x;
  int row = i >> 6;
  int col = i & 63;
  if (row < nrows) {
    float v = agg[(size_t)row * 64 + col] * di[row] + bias[col];
    out[(size_t)row * ldout + col] = fmaxf(v, 0.0f);
  }
}

// ---------------------------------------------------------------------------
// Host launcher
// ---------------------------------------------------------------------------
extern "C" void kernel_launch(void* const* d_in, const int* in_sizes, int n_in,
                              void* d_out, int out_size, void* d_ws, size_t ws_size,
                              hipStream_t stream) {
  const float* feat = (const float*)d_in[0];
  const int*   src  = (const int*)  d_in[1];
  const int*   dst  = (const int*)  d_in[2];
  const float* W1   = (const float*)d_in[3];
  const float* b1   = (const float*)d_in[4];
  const float* W2   = (const float*)d_in[5];
  const float* b2   = (const float*)d_in[6];
  const float* W3   = (const float*)d_in[7];
  const float* b3   = (const float*)d_in[8];
  const float* Wm   = (const float*)d_in[9];
  const float* bm   = (const float*)d_in[10];

  const int Nn = in_sizes[0] / 256;   // 100000
  const int E  = in_sizes[1];         // 1600000

  // Workspace layout (floats): do_isqrt[N] di_isqrt[N] hlin[N*64] agg[N*64]
  //                            hcat[N*192] aggjk[N*192]  then packed f16 W
  float* ws      = (float*)d_ws;
  float* deg     = ws;                               // do_isqrt then di_isqrt
  float* di_isq  = deg + Nn;
  float* hlin    = di_isq + Nn;
  float* agg     = hlin + (size_t)Nn * 64;
  float* hcat    = agg  + (size_t)Nn * 64;
  float* aggjk   = hcat + (size_t)Nn * 192;

  // Packed weight fragments (f16), each region 32B-aligned.
  // frag counts: W1: 8kt*4nt*32 = 1024, W2/W3: 2*4*32 = 256, Wm: 6*2*32 = 384
  _Float16* wp1 = (_Float16*)(aggjk + (size_t)Nn * 192);
  _Float16* wp2 = wp1 + (size_t)1024 * 16;
  _Float16* wp3 = wp2 + (size_t)256 * 16;
  _Float16* wpm = wp3 + (size_t)256 * 16;

  const int ZB = 2048;                               // grid-stride zero fill

  // 0) pack weights into WMMA B-fragment layout (tiny, L2 resident)
  pack_w_kernel<<<(1024 + 255) / 256, 256, 0, stream>>>(W1, 64, 8, 4, wp1);
  pack_w_kernel<<<1, 256, 0, stream>>>(W2, 64, 2, 4, wp2);
  pack_w_kernel<<<1, 256, 0, stream>>>(W3, 64, 2, 4, wp3);
  pack_w_kernel<<<2, 256, 0, stream>>>(Wm, 32, 6, 2, wpm);

  // 1) degrees -> deg^{-1/2}
  zero_kernel<<<ZB, 256, 0, stream>>>(deg, (long long)2 * Nn);
  degree_kernel<<<(E + 255) / 256, 256, 0, stream>>>(src, dst, deg, di_isq, E);
  isqrt_kernel<<<(2 * Nn + 255) / 256, 256, 0, stream>>>(deg, (long long)2 * Nn);

  const int gemm_blocks = ((Nn + 15) / 16 + 7) / 8;
  const int relu_blocks = (Nn * 64 + 255) / 256;
  const long long tot64 = (long long)E * 16;         // 64 feats / 4 per thread
  const int agg64_blocks = (int)((tot64 + 255) / 256);

  // 2) three GCN layers: gemm -> scatter-add -> relu(scale+bias) into hcat
  // layer 1: A = feat [N,256]
  gemm_wmma_kernel<4><<<gemm_blocks, 256, 0, stream>>>(
      feat, 256, deg, wp1, nullptr, hlin, 64, Nn, 8);
  zero_kernel<<<ZB, 256, 0, stream>>>(agg, (long long)Nn * 64);
  edge_agg_kernel<<<agg64_blocks, 256, 0, stream>>>(hlin, 64, 16, src, dst, agg, 64, tot64);
  relu_scale_bias_kernel<<<relu_blocks, 256, 0, stream>>>(agg, di_isq, b1, hcat + 0, 192, Nn);

  // layer 2: A = h1 = hcat[:, 0:64]
  gemm_wmma_kernel<4><<<gemm_blocks, 256, 0, stream>>>(
      hcat, 192, deg, wp2, nullptr, hlin, 64, Nn, 2);
  zero_kernel<<<ZB, 256, 0, stream>>>(agg, (long long)Nn * 64);
  edge_agg_kernel<<<agg64_blocks, 256, 0, stream>>>(hlin, 64, 16, src, dst, agg, 64, tot64);
  relu_scale_bias_kernel<<<relu_blocks, 256, 0, stream>>>(agg, di_isq, b2, hcat + 64, 192, Nn);

  // layer 3: A = h2 = hcat[:, 64:128]
  gemm_wmma_kernel<4><<<gemm_blocks, 256, 0, stream>>>(
      hcat + 64, 192, deg, wp3, nullptr, hlin, 64, Nn, 2);
  zero_kernel<<<ZB, 256, 0, stream>>>(agg, (long long)Nn * 64);
  edge_agg_kernel<<<agg64_blocks, 256, 0, stream>>>(hlin, 64, 16, src, dst, agg, 64, tot64);
  relu_scale_bias_kernel<<<relu_blocks, 256, 0, stream>>>(agg, di_isq, b3, hcat + 128, 192, Nn);

  // 3) JK aggregation over concat [N,192] (un-normalized copy_u/sum)
  zero_kernel<<<ZB, 256, 0, stream>>>(aggjk, (long long)Nn * 192);
  const long long tot192 = (long long)E * 48;
  edge_agg_kernel<<<(int)((tot192 + 255) / 256), 256, 0, stream>>>(
      hcat, 192, 48, src, dst, aggjk, 192, tot192);

  // 4) final projection: out = aggjk @ Wm + bm  -> d_out [N,32] f32
  gemm_wmma_kernel<2><<<gemm_blocks, 256, 0, stream>>>(
      aggjk, 192, nullptr, wpm, bm, (float*)d_out, 32, Nn, 6);
}
